// SparsTriangularUpdate_83502754168916
// MI455X (gfx1250) — compile-verified, matrix operand order checked by reading, not run on hardware
//
#include <hip/hip_runtime.h>
#include <cstdint>

#define NN 4096
#define KK 12
#define DD 256
#define CC 128
#define MM (NN * KK)

typedef __attribute__((ext_vector_type(16))) _Float16 v16h;
typedef __attribute__((ext_vector_type(8)))  _Float16 v8h;
typedef __attribute__((ext_vector_type(8)))  float    v8f;

__device__ __forceinline__ float wave_reduce_sum(float v) {
#pragma unroll
  for (int m = 16; m >= 1; m >>= 1) v += __shfl_xor(v, m, 32);
  return v;
}

__device__ __forceinline__ float sigmoidf_(float x) {
  return 1.0f / (1.0f + __expf(-x));
}

// Load a 16x32 f16 WMMA A/B fragment (wave32 layout, ISA 7.12.2):
// lanes 0-15 take K=[0..7,16..23], lanes 16-31 take K=[8..15,24..31]; within a lane,
// VGPRs 0-3 hold the first 8 K values, VGPRs 4-7 the +16 chunk.
// `base` must already include (row*ld + k0 + (lane>>4)*8).
__device__ __forceinline__ v16h load_ab_frag(const _Float16* base) {
  v8h lo = *(const v8h*)(base);
  v8h hi = *(const v8h*)(base + 16);
  v16h r;
#pragma unroll
  for (int i = 0; i < 8; ++i) { r[i] = lo[i]; r[8 + i] = hi[i]; }
  return r;
}

// ---------------- Stage 0: weight transpose f32 -> f16 (N-major) ----------
__global__ void transpose_w_kernel(const float* __restrict__ W,
                                   _Float16* __restrict__ Wt,
                                   int rows, int cols) {
  int t = blockIdx.x * 256 + threadIdx.x;
  if (t < rows * cols) {
    int i = t / cols;       // K index
    int j = t - i * cols;   // N index
    Wt[(size_t)j * rows + i] = (_Float16)W[t];
  }
}

// ---------------- Stage 1/4a: LayerNorm (wave per row) --------------------
template <int COLS>
__global__ void ln_kernel(const float* __restrict__ x,
                          const float* __restrict__ g,
                          const float* __restrict__ bt,
                          _Float16* __restrict__ out) {
  const int lane = threadIdx.x & 31;
  const int row  = blockIdx.x * 8 + (threadIdx.x >> 5);
  const float* xr = x + (size_t)row * COLS;
  constexpr int PL = COLS / 32;
  float v[PL];
  float s = 0.f;
#pragma unroll
  for (int i = 0; i < PL; ++i) { v[i] = xr[lane + 32 * i]; s += v[i]; }
  s = wave_reduce_sum(s);
  const float mu = s * (1.0f / COLS);
  float q = 0.f;
#pragma unroll
  for (int i = 0; i < PL; ++i) { float d = v[i] - mu; q += d * d; }
  q = wave_reduce_sum(q);
  const float rs = rsqrtf(q * (1.0f / COLS) + 1e-5f);
  _Float16* orow = out + (size_t)row * COLS;
#pragma unroll
  for (int i = 0; i < PL; ++i) {
    const int c = lane + 32 * i;
    orow[c] = (_Float16)((v[i] - mu) * rs * g[c] + bt[c]);
  }
}

// ------------- Stage 2: fused gated projection GEMM -----------------------
// Grid: NN/16 blocks, 256 threads (8 waves).
// Wave w: pair = w&1 (0 -> a via W_la/W_ga, 1 -> b via W_lb/W_gb),
//         columns [(w>>1)*32, +32) as two 16-wide N tiles.
// Accumulators: 2 weights x 2 tiles x 8 VGPR = 32 VGPRs (no spills).
__global__ void proj_gemm_kernel(const _Float16* __restrict__ xn,
                                 const _Float16* __restrict__ WtLa,
                                 const _Float16* __restrict__ WtGa,
                                 const _Float16* __restrict__ WtLb,
                                 const _Float16* __restrict__ WtGb,
                                 const float* __restrict__ bLa,
                                 const float* __restrict__ bGa,
                                 const float* __restrict__ bLb,
                                 const float* __restrict__ bGb,
                                 float* __restrict__ aOut,
                                 float* __restrict__ bOut) {
  const int lane = threadIdx.x & 31;
  const int wave = threadIdx.x >> 5;
  const int pair = wave & 1;
  const int cql  = wave >> 1;          // 0..3 -> 32-column slice
  const int half = lane >> 4;
  const int r    = lane & 15;
  const int m0   = blockIdx.x * 16;

  const _Float16* Wl = pair ? WtLb : WtLa;
  const _Float16* Wg = pair ? WtGb : WtGa;
  const float*    bl = pair ? bLb : bLa;
  const float*    bg = pair ? bGb : bGa;
  float*          dst = pair ? bOut : aOut;

  v8f accL[2], accG[2];
#pragma unroll
  for (int t = 0; t < 2; ++t) {
    accL[t] = (v8f){0.f, 0.f, 0.f, 0.f, 0.f, 0.f, 0.f, 0.f};
    accG[t] = (v8f){0.f, 0.f, 0.f, 0.f, 0.f, 0.f, 0.f, 0.f};
  }

  for (int k0 = 0; k0 < DD; k0 += 32) {
    const v16h afrag = load_ab_frag(xn + (size_t)(m0 + r) * DD + k0 + half * 8);
#pragma unroll
    for (int t = 0; t < 2; ++t) {
      const int n = cql * 32 + t * 16 + r;
      const v16h bl_frag = load_ab_frag(Wl + (size_t)n * DD + k0 + half * 8);
      accL[t] = __builtin_amdgcn_wmma_f32_16x16x32_f16(
          false, afrag, false, bl_frag, (short)0, accL[t], false, false);
      const v16h bg_frag = load_ab_frag(Wg + (size_t)n * DD + k0 + half * 8);
      accG[t] = __builtin_amdgcn_wmma_f32_16x16x32_f16(
          false, afrag, false, bg_frag, (short)0, accG[t], false, false);
    }
  }

#pragma unroll
  for (int t = 0; t < 2; ++t) {
    const int n = cql * 32 + t * 16 + r;
    const float vl = bl[n], vg = bg[n];
#pragma unroll
    for (int rr = 0; rr < 8; ++rr) {
      const int m = m0 + half * 8 + rr;
      dst[(size_t)m * CC + n] = sigmoidf_(accG[t][rr] + vg) * (accL[t][rr] + vl);
    }
  }
}

// ------------- Stage 3: sparse pair-matching einsum -----------------------
// One wave per (n,d). Matching logic is uniform across the wave; lanes
// parallelize the 128 channels.
__global__ void pair_kernel(const float* __restrict__ a,
                            const float* __restrict__ b,
                            const int* __restrict__ neighbors,
                            const int* __restrict__ edge_ids,
                            float* __restrict__ kout) {
  const int lane = threadIdx.x & 31;
  const int wid  = blockIdx.x * 8 + (threadIdx.x >> 5);
  const int n = wid / KK;
  const int d = wid - n * KK;

  int ie[KK + 1];
#pragma unroll
  for (int p = 0; p < KK; ++p) ie[p] = neighbors[n * KK + p];
  ie[KK] = n;
  const int nr = ie[d];
  int nj[KK];
#pragma unroll
  for (int q = 0; q < KK; ++q) nj[q] = neighbors[nr * KK + q];

  int pl[KK + 1], ql[KK];
  int ci = 0, cj = 0;
#pragma unroll
  for (int p = 0; p < KK + 1; ++p) {
    bool m = false;
#pragma unroll
    for (int q = 0; q < KK; ++q) m = m || (ie[p] == nj[q]);
    if (m) pl[ci++] = p;
  }
#pragma unroll
  for (int q = 0; q < KK; ++q) {
    bool m = false;
#pragma unroll
    for (int p = 0; p < KK + 1; ++p) m = m || (nj[q] == ie[p]);
    if (m) ql[cj++] = q;
  }
  const int np = ci < cj ? ci : cj;

  float acc[4] = {0.f, 0.f, 0.f, 0.f};
  for (int rr = 0; rr < np; ++rr) {
    const float* ar = a + (size_t)ie[pl[rr]] * CC;
    const float* br = b + (size_t)nj[ql[rr]] * CC;
#pragma unroll
    for (int j = 0; j < 4; ++j) acc[j] += ar[lane + 32 * j] * br[lane + 32 * j];
  }
  const int orow = edge_ids[n * KK + d];
  float* kr = kout + (size_t)orow * CC;
#pragma unroll
  for (int j = 0; j < 4; ++j) kr[lane + 32 * j] = acc[j];
}

// ------------- Stage 4b: fused output GEMMs + gating ----------------------
// Grid: MM/16 blocks, 256 threads (8 waves). Wave w owns cols [w*32,w*32+32).
__global__ void out_gemm_kernel(const _Float16* __restrict__ xn,
                                const _Float16* __restrict__ kn,
                                const _Float16* __restrict__ WtGo,
                                const _Float16* __restrict__ WtLo,
                                const float* __restrict__ bGo,
                                const float* __restrict__ bLo,
                                float* __restrict__ out) {
  const int lane = threadIdx.x & 31;
  const int wave = threadIdx.x >> 5;
  const int half = lane >> 4;
  const int r    = lane & 15;
  const int m0   = blockIdx.x * 16;

  v8f accG[2], accL[2];
#pragma unroll
  for (int t = 0; t < 2; ++t) {
    accG[t] = (v8f){0.f, 0.f, 0.f, 0.f, 0.f, 0.f, 0.f, 0.f};
    accL[t] = (v8f){0.f, 0.f, 0.f, 0.f, 0.f, 0.f, 0.f, 0.f};
  }

  // Gate path: xn (M x 256) @ W_go (256 x 256)
  for (int k0 = 0; k0 < DD; k0 += 32) {
    const v16h afrag = load_ab_frag(xn + (size_t)(m0 + r) * DD + k0 + half * 8);
#pragma unroll
    for (int t = 0; t < 2; ++t) {
      const int nb = wave * 32 + t * 16 + r;
      const v16h bfrag = load_ab_frag(WtGo + (size_t)nb * DD + k0 + half * 8);
      accG[t] = __builtin_amdgcn_wmma_f32_16x16x32_f16(
          false, afrag, false, bfrag, (short)0, accG[t], false, false);
    }
  }
  // Linear path: kn (M x 128) @ W_lo (128 x 256)
  for (int k0 = 0; k0 < CC; k0 += 32) {
    const v16h afrag = load_ab_frag(kn + (size_t)(m0 + r) * CC + k0 + half * 8);
#pragma unroll
    for (int t = 0; t < 2; ++t) {
      const int nb = wave * 32 + t * 16 + r;
      const v16h bfrag = load_ab_frag(WtLo + (size_t)nb * CC + k0 + half * 8);
      accL[t] = __builtin_amdgcn_wmma_f32_16x16x32_f16(
          false, afrag, false, bfrag, (short)0, accL[t], false, false);
    }
  }

#pragma unroll
  for (int t = 0; t < 2; ++t) {
    const int n = wave * 32 + t * 16 + r;
    const float vg = bGo[n], vl = bLo[n];
#pragma unroll
    for (int rr = 0; rr < 8; ++rr) {
      const int m = m0 + half * 8 + rr;
      out[(size_t)m * DD + n] = sigmoidf_(accG[t][rr] + vg) * (accL[t][rr] + vl);
    }
  }
}

extern "C" void kernel_launch(void* const* d_in, const int* in_sizes, int n_in,
                              void* d_out, int out_size, void* d_ws, size_t ws_size,
                              hipStream_t stream) {
  (void)in_sizes; (void)n_in; (void)out_size; (void)ws_size;

  const float* x        = (const float*)d_in[0];
  const float* W_la     = (const float*)d_in[1];
  const float* b_la     = (const float*)d_in[2];
  const float* W_ga     = (const float*)d_in[3];
  const float* b_ga     = (const float*)d_in[4];
  const float* W_lb     = (const float*)d_in[5];
  const float* b_lb     = (const float*)d_in[6];
  const float* W_gb     = (const float*)d_in[7];
  const float* b_gb     = (const float*)d_in[8];
  const float* W_go     = (const float*)d_in[9];
  const float* b_go     = (const float*)d_in[10];
  const float* W_lo     = (const float*)d_in[11];
  const float* b_lo     = (const float*)d_in[12];
  const float* gamma_in = (const float*)d_in[13];
  const float* beta_in  = (const float*)d_in[14];
  const float* gamma_o  = (const float*)d_in[15];
  const float* beta_o   = (const float*)d_in[16];
  const int* neighbors  = (const int*)d_in[17];
  const int* edge_ids   = (const int*)d_in[18];
  float* out = (float*)d_out;

  char* ws = (char*)d_ws;
  _Float16* xn = (_Float16*)ws;  ws += (size_t)MM * DD * sizeof(_Float16);
  float* aBuf  = (float*)ws;     ws += (size_t)NN * CC * sizeof(float);
  float* bBuf  = (float*)ws;     ws += (size_t)NN * CC * sizeof(float);
  float* kBuf  = (float*)ws;     ws += (size_t)MM * CC * sizeof(float);
  _Float16* kn = (_Float16*)ws;  ws += (size_t)MM * CC * sizeof(_Float16);
  _Float16* WtLa = (_Float16*)ws; ws += (size_t)CC * DD * sizeof(_Float16);
  _Float16* WtGa = (_Float16*)ws; ws += (size_t)CC * DD * sizeof(_Float16);
  _Float16* WtLb = (_Float16*)ws; ws += (size_t)CC * DD * sizeof(_Float16);
  _Float16* WtGb = (_Float16*)ws; ws += (size_t)CC * DD * sizeof(_Float16);
  _Float16* WtGo = (_Float16*)ws; ws += (size_t)DD * DD * sizeof(_Float16);
  _Float16* WtLo = (_Float16*)ws; ws += (size_t)DD * CC * sizeof(_Float16);

  // Stage 0: transpose weights into f16 N-major layout.
  const int tdc = (DD * CC + 255) / 256;
  transpose_w_kernel<<<tdc, 256, 0, stream>>>(W_la, WtLa, DD, CC);
  transpose_w_kernel<<<tdc, 256, 0, stream>>>(W_ga, WtGa, DD, CC);
  transpose_w_kernel<<<tdc, 256, 0, stream>>>(W_lb, WtLb, DD, CC);
  transpose_w_kernel<<<tdc, 256, 0, stream>>>(W_gb, WtGb, DD, CC);
  transpose_w_kernel<<<(DD * DD + 255) / 256, 256, 0, stream>>>(W_go, WtGo, DD, DD);
  transpose_w_kernel<<<(CC * DD + 255) / 256, 256, 0, stream>>>(W_lo, WtLo, CC, DD);

  // Stage 1: LayerNorm(x) -> xn (f16), all M rows.
  ln_kernel<DD><<<MM / 8, 256, 0, stream>>>(x, gamma_in, beta_in, xn);

  // Stage 2: gated projections a,b — only first N rows are ever consumed.
  proj_gemm_kernel<<<NN / 16, 256, 0, stream>>>(xn, WtLa, WtGa, WtLb, WtGb,
                                                b_la, b_ga, b_lb, b_gb,
                                                aBuf, bBuf);

  // Stage 3: sparse pair einsum -> k (M x C).
  pair_kernel<<<(NN * KK) / 8, 256, 0, stream>>>(aBuf, bBuf, neighbors,
                                                 edge_ids, kBuf);

  // Stage 4a: LayerNorm(k) -> kn (f16).
  ln_kernel<CC><<<MM / 8, 256, 0, stream>>>(kBuf, gamma_o, beta_o, kn);

  // Stage 4b: gated output GEMMs -> out (f32).
  out_gemm_kernel<<<MM / 16, 256, 0, stream>>>(xn, kn, WtGo, WtLo,
                                               b_go, b_lo, out);
}
